// PointNetFeaturePropagation_30468497997878
// MI455X (gfx1250) — compile-verified, matrix operand order checked by reading, not run on hardware
//
#include <hip/hip_runtime.h>
#include <hip/hip_bf16.h>

// ---------------- problem constants ----------------
#define BB 16
#define NN 4096
#define SS 1024
#define C1 128
#define C2 256
#define CIN 384   // C1 + C2
#define COUT 256

typedef __attribute__((ext_vector_type(16))) __bf16 v16bf;
typedef __attribute__((ext_vector_type(8)))  float  v8f;

union Frag {
    uint4 u4[2];
    v16bf v;
};

// CDNA5 async global->LDS copy (ASYNCcnt-tracked, no VGPR staging).
// VDST VGPR holds the LDS byte offset (low 32 bits of generic shared addr,
// per flat-aperture truncation rule); VADDR holds the 64-bit global address.
__device__ __forceinline__ void async_copy_b128(const void* gptr, void* lptr) {
    unsigned lofs = (unsigned)(uintptr_t)lptr;
    unsigned long long ga = (unsigned long long)(uintptr_t)gptr;
    asm volatile("global_load_async_to_lds_b128 %0, %1, off"
                 :: "v"(lofs), "v"(ga)
                 : "memory");
}
__device__ __forceinline__ void wait_async0() {
    asm volatile("s_wait_asynccnt 0x0" ::: "memory");
}

// ---------------------------------------------------------------------------
// Kernel 1: convert + transpose weights to bf16, Wt[n][k] layout (K contiguous)
// ---------------------------------------------------------------------------
__global__ void prep_w_kernel(const float* __restrict__ w0,
                              const float* __restrict__ w1,
                              __bf16* __restrict__ Wt0,
                              __bf16* __restrict__ Wt1) {
    int i = blockIdx.x * 256 + threadIdx.x;
    if (i < CIN * COUT) {
        int k = i / COUT, n = i % COUT;          // w0 is [CIN, COUT] row-major
        Wt0[n * CIN + k] = (__bf16)w0[i];
    } else {
        int j = i - CIN * COUT;
        if (j < COUT * COUT) {
            int k = j / COUT, n = j % COUT;
            Wt1[n * COUT + k] = (__bf16)w1[j];
        }
    }
}

// ---------------------------------------------------------------------------
// Kernel 2: three_nn + three_interpolate + concat -> A0 bf16 [B*N, 384]
// One block (256 threads) handles 256 points of one batch.
// ---------------------------------------------------------------------------
__global__ void three_nn_interp_kernel(const float* __restrict__ xyz1,
                                       const float* __restrict__ xyz2,
                                       const float* __restrict__ points1,
                                       const float* __restrict__ points2,
                                       __bf16* __restrict__ A0) {
    __shared__ float sx[SS];
    __shared__ float sy[SS];
    __shared__ float sz[SS];
    __shared__ int   sidx[256][3];
    __shared__ float swgt[256][3];

    const int tid    = threadIdx.x;
    const int b      = blockIdx.x >> 4;          // 16 blocks per batch
    const int n_base = (blockIdx.x & 15) << 8;   // 256 points per block

    // stage this batch's xyz2 into LDS (deinterleave xyz)
    const float* x2 = xyz2 + (size_t)b * SS * 3;
    for (int i = tid; i < SS * 3; i += 256) {
        float v = x2[i];
        int p = i / 3, c = i - p * 3;
        if (c == 0) sx[p] = v;
        else if (c == 1) sy[p] = v;
        else sz[p] = v;
    }
    __syncthreads();

    // per-thread point: scan all S candidates, keep 3 smallest squared dists
    const int n = n_base + tid;
    const float* p1 = xyz1 + ((size_t)b * NN + n) * 3;
    const float px = p1[0], py = p1[1], pz = p1[2];

    float d0 = 3.0e38f, d1 = 3.0e38f, d2 = 3.0e38f;
    int   i0 = 0, i1 = 0, i2 = 0;
    for (int s = 0; s < SS; ++s) {
        float dx = px - sx[s];
        float dy = py - sy[s];
        float dz = pz - sz[s];
        float d = dx * dx;
        d = fmaf(dy, dy, d);
        d = fmaf(dz, dz, d);
        if (d < d0)      { d2 = d1; i2 = i1; d1 = d0; i1 = i0; d0 = d; i0 = s; }
        else if (d < d1) { d2 = d1; i2 = i1; d1 = d;  i1 = s; }
        else if (d < d2) { d2 = d;  i2 = s; }
    }
    d0 = fmaxf(d0, 1e-10f);
    d1 = fmaxf(d1, 1e-10f);
    d2 = fmaxf(d2, 1e-10f);
    float inv0 = 1.0f / d0, inv1 = 1.0f / d1, inv2 = 1.0f / d2;
    float rs = 1.0f / (inv0 + inv1 + inv2);
    sidx[tid][0] = i0;        sidx[tid][1] = i1;        sidx[tid][2] = i2;
    swgt[tid][0] = inv0 * rs; swgt[tid][1] = inv1 * rs; swgt[tid][2] = inv2 * rs;
    __syncthreads();

    // block-cooperative gather: coalesced over channels
    const float* P2 = points2 + (size_t)b * SS * C2;
    for (int p = 0; p < 256; ++p) {
        int   j0 = sidx[p][0], j1 = sidx[p][1], j2 = sidx[p][2];
        float w0 = swgt[p][0], w1 = swgt[p][1], w2 = swgt[p][2];
        size_t row = (size_t)b * NN + n_base + p;
        // interpolated C2 channels (cols 0..255)
        float v = w0 * P2[j0 * C2 + tid]
                + w1 * P2[j1 * C2 + tid]
                + w2 * P2[j2 * C2 + tid];
        A0[row * CIN + tid] = (__bf16)v;
        // points1 passthrough (cols 256..383)
        if (tid < C1) {
            A0[row * CIN + C2 + tid] = (__bf16)points1[row * C1 + tid];
        }
    }
}

// ---------------------------------------------------------------------------
// Kernel 3/4: bf16 GEMM + bias + relu using v_wmma_f32_16x16x32_bf16.
//   A  : [M, K]   bf16 row-major
//   Wt : [256, K] bf16 (output-channel major, K contiguous)
//   out: [M, 256] bf16 or f32
// Block: 256 threads = 8 waves, tile 128x128, wave tile 32x64 (2x4 wmma).
// Double-buffered LDS fed by global_load_async_to_lds_b128 (ASYNCcnt).
// ---------------------------------------------------------------------------
#define KC  64
#define LDP 72   // padded LDS row (bank-conflict relief), 144B, 16B aligned

__device__ __forceinline__ void stage_chunk(const __bf16* __restrict__ A,
                                            const __bf16* __restrict__ Wt,
                                            int row0, int col0, int K, int kc,
                                            __bf16 (*As)[LDP], __bf16 (*Bs)[LDP],
                                            int tid) {
#pragma unroll
    for (int i = 0; i < 4; ++i) {
        int linear = tid + i * 256;           // 1024 16B vectors per matrix
        int r = linear >> 3;
        int seg = (linear & 7) << 3;          // bf16 column offset
        async_copy_b128(A  + (size_t)(row0 + r) * K + kc + seg, &As[r][seg]);
        async_copy_b128(Wt + (size_t)(col0 + r) * K + kc + seg, &Bs[r][seg]);
    }
}

template <bool OUT_BF16>
__global__ void gemm_bias_relu_kernel(const __bf16* __restrict__ A,
                                      const __bf16* __restrict__ Wt,
                                      const float* __restrict__ bias,
                                      void* __restrict__ outp,
                                      int M, int K) {
    __shared__ __align__(16) __bf16 As[2][128][LDP];
    __shared__ __align__(16) __bf16 Bs[2][128][LDP];

    const int tid  = threadIdx.x;
    const int lane = tid & 31;
    const int wv   = tid >> 5;
    const int wr   = wv >> 1;                     // 0..3  -> row group (32 rows)
    const int wc   = wv & 1;                      // 0..1  -> col group (64 cols)
    const int l15  = lane & 15;
    const int lhf  = lane >> 4;                   // lane half

    const int row0 = blockIdx.x * 128;
    const int col0 = blockIdx.y * 128;

    const v8f vzero = {0.f, 0.f, 0.f, 0.f, 0.f, 0.f, 0.f, 0.f};
    v8f acc[2][4];
#pragma unroll
    for (int i = 0; i < 2; ++i)
#pragma unroll
        for (int j = 0; j < 4; ++j) acc[i][j] = vzero;

    const int nchunks = K / KC;

    // prologue: async-stage chunk 0 into buffer 0
    stage_chunk(A, Wt, row0, col0, K, 0, As[0], Bs[0], tid);
    wait_async0();
    __syncthreads();

    for (int c = 0; c < nchunks; ++c) {
        const int cur = c & 1;
        // overlap: issue async loads of next chunk into the other buffer
        if (c + 1 < nchunks)
            stage_chunk(A, Wt, row0, col0, K, (c + 1) * KC,
                        As[cur ^ 1], Bs[cur ^ 1], tid);

#pragma unroll
        for (int kk = 0; kk < KC; kk += 32) {
            Frag fa[2], fb[4];
#pragma unroll
            for (int i = 0; i < 2; ++i) {
                int m = wr * 32 + i * 16 + l15;
                fa[i].u4[0] = *(const uint4*)&As[cur][m][kk + lhf * 8];
                fa[i].u4[1] = *(const uint4*)&As[cur][m][kk + 16 + lhf * 8];
            }
#pragma unroll
            for (int j = 0; j < 4; ++j) {
                int nn2 = wc * 64 + j * 16 + l15;
                fb[j].u4[0] = *(const uint4*)&Bs[cur][nn2][kk + lhf * 8];
                fb[j].u4[1] = *(const uint4*)&Bs[cur][nn2][kk + 16 + lhf * 8];
            }
#pragma unroll
            for (int i = 0; i < 2; ++i)
#pragma unroll
                for (int j = 0; j < 4; ++j)
                    acc[i][j] = __builtin_amdgcn_wmma_f32_16x16x32_bf16(
                        false, fa[i].v, false, fb[j].v,
                        (short)0, acc[i][j], false, false);
        }

        // make next buffer's async writes visible to all waves
        if (c + 1 < nchunks)
            wait_async0();
        __syncthreads();
    }

    // epilogue: bias + relu, scatter per C-matrix layout
#pragma unroll
    for (int i = 0; i < 2; ++i) {
#pragma unroll
        for (int j = 0; j < 4; ++j) {
            int nglob = col0 + wc * 64 + j * 16 + l15;
            float bv = bias[nglob];
#pragma unroll
            for (int v = 0; v < 8; ++v) {
                int mglob = row0 + wr * 32 + i * 16 + lhf * 8 + v;
                float val = acc[i][j][v] + bv;
                val = val > 0.f ? val : 0.f;
                if (OUT_BF16)
                    ((__bf16*)outp)[(size_t)mglob * COUT + nglob] = (__bf16)val;
                else
                    ((float*)outp)[(size_t)mglob * COUT + nglob] = val;
            }
        }
    }
}

// ---------------------------------------------------------------------------
extern "C" void kernel_launch(void* const* d_in, const int* in_sizes, int n_in,
                              void* d_out, int out_size, void* d_ws, size_t ws_size,
                              hipStream_t stream) {
    const float* xyz1    = (const float*)d_in[0];
    const float* xyz2    = (const float*)d_in[1];
    const float* points1 = (const float*)d_in[2];
    const float* points2 = (const float*)d_in[3];
    const float* w0      = (const float*)d_in[4];
    const float* b0      = (const float*)d_in[5];
    const float* w1      = (const float*)d_in[6];
    const float* b1      = (const float*)d_in[7];
    float* out = (float*)d_out;

    // workspace layout (bytes)
    char* ws = (char*)d_ws;
    const size_t off_wt0 = 0;
    const size_t off_wt1 = off_wt0 + (size_t)COUT * CIN  * 2;   // 196608
    const size_t off_a0  = off_wt1 + (size_t)COUT * COUT * 2;   // 327680
    const size_t off_h   = off_a0  + (size_t)BB * NN * CIN * 2; // +48MB
    __bf16* Wt0 = (__bf16*)(ws + off_wt0);
    __bf16* Wt1 = (__bf16*)(ws + off_wt1);
    __bf16* A0  = (__bf16*)(ws + off_a0);
    __bf16* H   = (__bf16*)(ws + off_h);

    // 1) weight transpose+convert
    {
        int total = CIN * COUT + COUT * COUT;
        prep_w_kernel<<<(total + 255) / 256, 256, 0, stream>>>(w0, w1, Wt0, Wt1);
    }
    // 2) three_nn + interpolate + concat -> A0 (bf16, GEMM-ready)
    {
        dim3 grid(BB * (NN / 256));
        three_nn_interp_kernel<<<grid, 256, 0, stream>>>(xyz1, xyz2, points1, points2, A0);
    }
    // 3) layer 0: [65536 x 384] x [384 x 256] -> H (bf16)
    {
        dim3 grid((BB * NN) / 128, COUT / 128);
        gemm_bias_relu_kernel<true><<<grid, 256, 0, stream>>>(A0, Wt0, b0, (void*)H,
                                                              BB * NN, CIN);
    }
    // 4) layer 1: [65536 x 256] x [256 x 256] -> out (f32)
    {
        dim3 grid((BB * NN) / 128, COUT / 128);
        gemm_bias_relu_kernel<false><<<grid, 256, 0, stream>>>(H, Wt1, b1, (void*)out,
                                                               BB * NN, COUT);
    }
}
